// KVCachedDecoder_90245852823568
// MI455X (gfx1250) — matrix operand order repaired and checked by reading, not compiled, not run on hardware
//
#include <hip/hip_runtime.h>
#include <math.h>

// ---------------- problem constants (match reference) ----------------
constexpr int L_ = 3;
constexpr int B_ = 16;
constexpr int D_ = 1024;
constexpr int H_ = 16;
constexpr int HD_ = 64;
constexpr int T_PAST_ = 1023;
constexpr int S_ = 1024;      // S_ENC, also total T for self-attn (1023+1)
constexpr int FF_ = 4096;
constexpr int V_ = 20;
constexpr float EPS_ = 1e-5f;
constexpr float SCALE_ = 0.125f;  // 1/sqrt(64)

typedef __attribute__((ext_vector_type(16))) __bf16 v16bf;
typedef __attribute__((ext_vector_type(2)))  __bf16 v2bf;
typedef __attribute__((ext_vector_type(8)))  float  v8f;

// =====================================================================
// GEMM: C[M,N] = A[M,K] @ W[K,N] + bias (+optional relu)
// f32 in HBM -> bf16 fragments staged in LDS *already in WMMA register
// layout*, so each consumer fragment is one contiguous 32B LDS read
// (2x ds_load_b128). Block = 128 threads = 4 wave32s; block tile is
// (16*MT) x 64; each wave owns one 16-col slice and MT accumulators
// (B fragment reused across MT WMMAs).
// Requires: M % (16*MT) == 0, N % 64 == 0, K % 32 == 0.
// =====================================================================
template <int MT>
__global__ __launch_bounds__(128) void gemm_wmma_kernel(
    const float* __restrict__ A, int lda,
    const float* __restrict__ W, int ldw,
    const float* __restrict__ bias,
    float* __restrict__ C, int ldc,
    int K, int relu)
{
  // fragment-layout staging: [tile][lane][element], 32B per lane row
  __shared__ alignas(32) __bf16 sAfrag[MT][32][16];
  __shared__ alignas(32) __bf16 sBfrag[4][32][16];

  const int tid  = threadIdx.x;
  const int wave = tid >> 5;
  const int lane = tid & 31;
  const int m0   = blockIdx.y * (16 * MT);
  const int n0   = blockIdx.x * 64;

  const float* __restrict__ Ap = A + (size_t)m0 * lda;  // + k0 added per step
  const float* __restrict__ Wp = W + n0;                // + k0*ldw per step

  v8f acc[MT];
#pragma unroll
  for (int mt = 0; mt < MT; ++mt) acc[mt] = (v8f){0.f,0.f,0.f,0.f,0.f,0.f,0.f,0.f};

  for (int k0 = 0; k0 < K; k0 += 32) {
    // ---- batched global loads into registers (max MLP) ----
    // A: (16*MT) rows x 32 k  => MT*256 float2 pairs, MT*2 per thread
    float2 av[MT * 2];
#pragma unroll
    for (int j = 0; j < MT * 2; ++j) {
      int pid = tid + j * 128;
      int m = pid >> 4, pp = pid & 15;           // k = 2*pp
      av[j] = *(const float2*)(Ap + (size_t)m * lda + k0 + 2 * pp);
    }
    // W: 32 k x 64 n => 16 k-pairs x 64 n = 1024 pairs, 8 per thread
    float w0[8], w1[8];
#pragma unroll
    for (int j = 0; j < 8; ++j) {
      int pid = tid + j * 128;
      int n = pid & 63, kp = pid >> 6;           // k = 2*kp
      const float* wp = Wp + (size_t)(k0 + 2 * kp) * ldw + n;
      w0[j] = wp[0];
      w1[j] = wp[ldw];
    }
    // hint-prefetch next K-slab of W (global_prefetch_b8)
    if (k0 + 32 < K)
      __builtin_prefetch(Wp + (size_t)(k0 + 32 + (tid >> 3)) * ldw + ((tid & 7) << 3), 0, 1);

    // ---- convert + store in fragment layout (paired -> ds_store_b32) ----
    // A element map: lane = (m&15) + ((k&8)?16:0); e = (k&7) + ((k&16)?8:0)
#pragma unroll
    for (int j = 0; j < MT * 2; ++j) {
      int pid = tid + j * 128;
      int m = pid >> 4, k = (pid & 15) * 2;
      int ln = (m & 15) + ((k & 8) ? 16 : 0);
      int e  = (k & 7) + ((k & 16) ? 8 : 0);
      v2bf p; p.x = (__bf16)av[j].x; p.y = (__bf16)av[j].y;
      *(v2bf*)&sAfrag[m >> 4][ln][e] = p;
    }
    // B element map: lane = (n&15) + ((k&8)?16:0); e = (k&7) + ((k&16)?8:0)
#pragma unroll
    for (int j = 0; j < 8; ++j) {
      int pid = tid + j * 128;
      int n = pid & 63, k = (pid >> 6) * 2;
      int ln = (n & 15) + ((k & 8) ? 16 : 0);
      int e  = (k & 7) + ((k & 16) ? 8 : 0);
      v2bf p; p.x = (__bf16)w0[j]; p.y = (__bf16)w1[j];
      *(v2bf*)&sBfrag[n >> 4][ln][e] = p;
    }
    __syncthreads();

    // ---- fragments: one contiguous 32B LDS read each ----
    v16bf bfrag = *(const v16bf*)&sBfrag[wave][lane][0];
#pragma unroll
    for (int mt = 0; mt < MT; ++mt) {
      v16bf afrag = *(const v16bf*)&sAfrag[mt][lane][0];
      acc[mt] = __builtin_amdgcn_wmma_f32_16x16x32_bf16(
          /*neg_a=*/false, afrag, /*neg_b=*/false, bfrag,
          /*c_mod=*/(short)0, acc[mt], /*reuse_a=*/false, /*reuse_b=*/false);
    }
    __syncthreads();
  }

  // C/D layout: VGPR r -> M = r (lanes 0-15) or 8+r (lanes 16-31); N = lane&15
  const int nl  = lane & 15;
  const int mb  = (lane < 16) ? 0 : 8;
  const int ncg = n0 + wave * 16 + nl;
  const float bv = bias[ncg];
#pragma unroll
  for (int mt = 0; mt < MT; ++mt) {
#pragma unroll
    for (int r = 0; r < 8; ++r) {
      float v = acc[mt][r] + bv;
      if (relu) v = fmaxf(v, 0.f);
      C[(size_t)(m0 + mt * 16 + mb + r) * ldc + ncg] = v;
    }
  }
}

// =====================================================================
// Fused single-query attention: out[b, h*64+d] = softmax(q.K^T/8) @ V
// mode 0: K/V rows 0..1022 from cache [B,H,1023,64], row 1023 from knew/vnew
// mode 1: K/V rows from projected enc [B,S,D] (head column slice)
// One block per (b,h): 256 threads, T=1024.
// =====================================================================
__global__ __launch_bounds__(256) void attn1q_kernel(
    const float* __restrict__ qb,
    const float* __restrict__ kc, const float* __restrict__ vc,
    const float* __restrict__ knew, const float* __restrict__ vnew,
    float* __restrict__ out, int mode)
{
  __shared__ float sq[HD_];
  __shared__ float sc[1024];
  __shared__ float red[256];
  __shared__ float pout[4][HD_];

  const int bh = blockIdx.x;
  const int b  = bh / H_;
  const int h  = bh % H_;
  const int tid = threadIdx.x;

  if (tid < HD_) sq[tid] = qb[(size_t)b * D_ + h * HD_ + tid];
  __syncthreads();

  float loc[4];
#pragma unroll
  for (int c = 0; c < 4; ++c) {
    int t = tid + c * 256;
    const float* kr;
    if (mode == 0)
      kr = (t < T_PAST_) ? (kc + ((size_t)(b * H_ + h) * T_PAST_ + t) * HD_)
                         : (knew + (size_t)b * D_ + h * HD_);
    else
      kr = kc + ((size_t)b * S_ + t) * D_ + h * HD_;
    const float4* k4 = (const float4*)kr;
    float s = 0.f;
#pragma unroll
    for (int j = 0; j < 16; ++j) {
      float4 kv = k4[j];
      s += kv.x * sq[4 * j] + kv.y * sq[4 * j + 1] + kv.z * sq[4 * j + 2] + kv.w * sq[4 * j + 3];
    }
    loc[c] = s * SCALE_;
    sc[t] = loc[c];
  }
  float m = fmaxf(fmaxf(loc[0], loc[1]), fmaxf(loc[2], loc[3]));
  red[tid] = m; __syncthreads();
  for (int o = 128; o > 0; o >>= 1) {
    if (tid < o) red[tid] = fmaxf(red[tid], red[tid + o]);
    __syncthreads();
  }
  const float gmax = red[0];
  __syncthreads();
  float lsum = 0.f;
#pragma unroll
  for (int c = 0; c < 4; ++c) {
    int t = tid + c * 256;
    float e = __expf(sc[t] - gmax);
    sc[t] = e;
    lsum += e;
  }
  red[tid] = lsum; __syncthreads();
  for (int o = 128; o > 0; o >>= 1) {
    if (tid < o) red[tid] += red[tid + o];
    __syncthreads();
  }
  const float inv = 1.f / red[0];
  __syncthreads();
  const int d = tid & 63, chunk = tid >> 6;
  float av = 0.f;
  for (int t = chunk * 256; t < chunk * 256 + 256; ++t) {
    const float* vr;
    if (mode == 0)
      vr = (t < T_PAST_) ? (vc + ((size_t)(b * H_ + h) * T_PAST_ + t) * HD_)
                         : (vnew + (size_t)b * D_ + h * HD_);
    else
      vr = vc + ((size_t)b * S_ + t) * D_ + h * HD_;
    av += sc[t] * vr[d];
  }
  pout[chunk][d] = av;
  __syncthreads();
  if (tid < HD_) {
    out[(size_t)b * D_ + h * HD_ + tid] =
        (pout[0][tid] + pout[1][tid] + pout[2][tid] + pout[3][tid]) * inv;
  }
}

// =====================================================================
// x[b,:] = LayerNorm(x[b,:] + a[b,:]) * g + beta ; one block per b row
// =====================================================================
__global__ __launch_bounds__(256) void resid_ln_kernel(
    float* __restrict__ x, const float* __restrict__ a,
    const float* __restrict__ g, const float* __restrict__ beta)
{
  __shared__ float red[256];
  const int b = blockIdx.x;
  const int tid = threadIdx.x;
  float v[4];
  float s = 0.f;
#pragma unroll
  for (int j = 0; j < 4; ++j) {
    int idx = tid + j * 256;
    v[j] = x[(size_t)b * D_ + idx] + a[(size_t)b * D_ + idx];
    s += v[j];
  }
  red[tid] = s; __syncthreads();
  for (int o = 128; o > 0; o >>= 1) { if (tid < o) red[tid] += red[tid + o]; __syncthreads(); }
  const float mean = red[0] * (1.f / D_);
  __syncthreads();
  float sq = 0.f;
#pragma unroll
  for (int j = 0; j < 4; ++j) { float dlt = v[j] - mean; sq += dlt * dlt; }
  red[tid] = sq; __syncthreads();
  for (int o = 128; o > 0; o >>= 1) { if (tid < o) red[tid] += red[tid + o]; __syncthreads(); }
  const float rstd = rsqrtf(red[0] * (1.f / D_) + EPS_);
  __syncthreads();
#pragma unroll
  for (int j = 0; j < 4; ++j) {
    int idx = tid + j * 256;
    x[(size_t)b * D_ + idx] = (v[j] - mean) * rstd * g[idx] + beta[idx];
  }
}

// =====================================================================
// logits[b,v] = x[b,:] @ w_out[:,v] + b_out[v]
// =====================================================================
__global__ __launch_bounds__(256) void logits_kernel(
    const float* __restrict__ x, const float* __restrict__ wout,
    const float* __restrict__ bout, float* __restrict__ out)
{
  __shared__ float red[256];
  const int v = blockIdx.x, b = blockIdx.y;
  float s = 0.f;
  for (int k = threadIdx.x; k < D_; k += 256)
    s += x[(size_t)b * D_ + k] * wout[(size_t)k * V_ + v];
  red[threadIdx.x] = s; __syncthreads();
  for (int o = 128; o > 0; o >>= 1) {
    if (threadIdx.x < o) red[threadIdx.x] += red[threadIdx.x + o];
    __syncthreads();
  }
  if (threadIdx.x == 0) out[(size_t)b * V_ + v] = red[0] + bout[v];
}

// =====================================================================
extern "C" void kernel_launch(void* const* d_in, const int* in_sizes, int n_in,
                              void* d_out, int out_size, void* d_ws, size_t ws_size,
                              hipStream_t stream)
{
  (void)in_sizes; (void)n_in; (void)out_size; (void)ws_size;

  const float* x_in    = (const float*)d_in[0];
  const float* enc     = (const float*)d_in[1];
  const float* k_cache = (const float*)d_in[2];
  const float* v_cache = (const float*)d_in[3];
  const float* wq_s = (const float*)d_in[4];  const float* bq_s = (const float*)d_in[5];
  const float* wk_s = (const float*)d_in[6];  const float* bk_s = (const float*)d_in[7];
  const float* wv_s = (const float*)d_in[8];  const float* bv_s = (const float*)d_in[9];
  const float* wo_s = (const float*)d_in[10]; const float* bo_s = (const float*)d_in[11];
  const float* wq_c = (const float*)d_in[12]; const float* bq_c = (const float*)d_in[13];
  const float* wk_c = (const float*)d_in[14]; const float* bk_c = (const float*)d_in[15];
  const float* wv_c = (const float*)d_in[16]; const float* bv_c = (const float*)d_in[17];
  const float* wo_c = (const float*)d_in[18]; const float* bo_c = (const float*)d_in[19];
  const float* w1   = (const float*)d_in[20]; const float* b1   = (const float*)d_in[21];
  const float* w2   = (const float*)d_in[22]; const float* b2   = (const float*)d_in[23];
  const float* ln1g = (const float*)d_in[24]; const float* ln1b = (const float*)d_in[25];
  const float* ln2g = (const float*)d_in[26]; const float* ln2b = (const float*)d_in[27];
  const float* ln3g = (const float*)d_in[28]; const float* ln3b = (const float*)d_in[29];
  const float* wout = (const float*)d_in[30]; const float* bout = (const float*)d_in[31];

  float* ws    = (float*)d_ws;
  float* xbuf  = ws;                       // B*D
  float* tmp   = xbuf + B_ * D_;           // B*D
  float* qbuf  = tmp  + B_ * D_;           // B*D
  float* knew  = qbuf + B_ * D_;           // B*D
  float* vnew  = knew + B_ * D_;           // B*D
  float* attno = vnew + B_ * D_;           // B*D
  float* ffb   = attno + B_ * D_;          // B*FF
  float* kproj = ffb + (size_t)B_ * FF_;   // B*S*D (64 MB)
  float* vproj = kproj + (size_t)B_ * S_ * D_;  // B*S*D (64 MB)

  hipMemcpyAsync(xbuf, x_in, (size_t)B_ * D_ * sizeof(float),
                 hipMemcpyDeviceToDevice, stream);

  auto gemm = [&](const float* A, int lda, const float* Wt, int ldw,
                  const float* bias, float* Cp, int M, int N, int K, int relu) {
    if (M % 64 == 0) {
      dim3 grid(N / 64, M / 64);
      gemm_wmma_kernel<4><<<grid, 128, 0, stream>>>(A, lda, Wt, ldw, bias, Cp, N, K, relu);
    } else {
      dim3 grid(N / 64, M / 16);
      gemm_wmma_kernel<1><<<grid, 128, 0, stream>>>(A, lda, Wt, ldw, bias, Cp, N, K, relu);
    }
  };

  for (int l = 0; l < L_; ++l) {
    const size_t wOff = (size_t)l * D_ * D_;
    const size_t bOff = (size_t)l * D_;
    const size_t cOff = (size_t)l * B_ * H_ * T_PAST_ * HD_;

    // ---- self-attention (KV-cached, query length 1) ----
    gemm(xbuf, D_, wq_s + wOff, D_, bq_s + bOff, qbuf, B_, D_, D_, 0);
    gemm(xbuf, D_, wk_s + wOff, D_, bk_s + bOff, knew, B_, D_, D_, 0);
    gemm(xbuf, D_, wv_s + wOff, D_, bv_s + bOff, vnew, B_, D_, D_, 0);
    attn1q_kernel<<<dim3(B_ * H_), 256, 0, stream>>>(
        qbuf, k_cache + cOff, v_cache + cOff, knew, vnew, attno, /*mode=*/0);
    gemm(attno, D_, wo_s + wOff, D_, bo_s + bOff, tmp, B_, D_, D_, 0);
    resid_ln_kernel<<<dim3(B_), 256, 0, stream>>>(xbuf, tmp, ln1g + bOff, ln1b + bOff);

    // ---- cross-attention (enc K/V projections dominate the roofline) ----
    gemm(xbuf, D_, wq_c + wOff, D_, bq_c + bOff, qbuf, B_, D_, D_, 0);
    gemm(enc, D_, wk_c + wOff, D_, bk_c + bOff, kproj, B_ * S_, D_, D_, 0);
    gemm(enc, D_, wv_c + wOff, D_, bv_c + bOff, vproj, B_ * S_, D_, D_, 0);
    attn1q_kernel<<<dim3(B_ * H_), 256, 0, stream>>>(
        qbuf, kproj, vproj, nullptr, nullptr, attno, /*mode=*/1);
    gemm(attno, D_, wo_c + wOff, D_, bo_c + bOff, tmp, B_, D_, D_, 0);
    resid_ln_kernel<<<dim3(B_), 256, 0, stream>>>(xbuf, tmp, ln2g + bOff, ln2b + bOff);

    // ---- FFN ----
    gemm(xbuf, D_, w1 + (size_t)l * D_ * FF_, FF_, b1 + (size_t)l * FF_, ffb,
         B_, FF_, D_, /*relu=*/1);
    gemm(ffb, FF_, w2 + (size_t)l * FF_ * D_, D_, b2 + bOff, tmp, B_, D_, FF_, 0);
    resid_ln_kernel<<<dim3(B_), 256, 0, stream>>>(xbuf, tmp, ln3g + bOff, ln3b + bOff);
  }

  logits_kernel<<<dim3(V_, B_), 256, 0, stream>>>(xbuf, wout, bout, (float*)d_out);
}
